// VRNN_62397284876760
// MI455X (gfx1250) — matrix operand order repaired and checked by reading, not compiled, not run on hardware
//
#include <hip/hip_runtime.h>
#include <math.h>

typedef __bf16 bf16;
typedef __attribute__((ext_vector_type(16))) __bf16 v16bf;
typedef __attribute__((ext_vector_type(8)))  __bf16 v8bf;
typedef __attribute__((ext_vector_type(8)))  float  v8f;

#define T_STEPS  1000
#define BATCH    256
#define XDIM     128
#define HDIM     256
#define ZDIM     64
#define ROWS     16                 // batch rows per workgroup (one WMMA M-tile)
#define NWG      (BATCH / ROWS)     // 16 persistent workgroups
#define NTHREADS 256                // 8 wave32s
#define NWAVES   (NTHREADS / 32)

#define EPSF          1.1920929e-07f
#define LOG_2PI_HALF  0.91893853320467274f

// ---------------- dynamic LDS layout (bytes, all 16B aligned) ----------------
constexpr unsigned OFF_XF   = 0;                          // f32  16x128  x_t
constexpr unsigned OFF_XB   = OFF_XF   + ROWS*XDIM*4;     // bf16 16x128
constexpr unsigned OFF_HF   = OFF_XB   + ROWS*XDIM*2;     // f32  16x256  h
constexpr unsigned OFF_HB   = OFF_HF   + ROWS*HDIM*4;     // bf16 16x256
constexpr unsigned OFF_T0   = OFF_HB   + ROWS*HDIM*2;     // bf16 16x256  temp
constexpr unsigned OFF_PHIX = OFF_T0   + ROWS*HDIM*2;     // bf16 16x256
constexpr unsigned OFF_ENCH = OFF_PHIX + ROWS*HDIM*2;     // bf16 16x256
constexpr unsigned OFF_PRIH = OFF_ENCH + ROWS*HDIM*2;     // bf16 16x256
constexpr unsigned OFF_DECH = OFF_PRIH + ROWS*HDIM*2;     // bf16 16x256
constexpr unsigned OFF_PHIZ = OFF_DECH + ROWS*HDIM*2;     // bf16 16x256
constexpr unsigned OFF_ZB   = OFF_PHIZ + ROWS*HDIM*2;     // bf16 16x64
constexpr unsigned OFF_ENCM = OFF_ZB   + ROWS*ZDIM*2;     // f32  16x64
constexpr unsigned OFF_ENCS = OFF_ENCM + ROWS*ZDIM*4;     // f32  16x64
constexpr unsigned OFF_PRIM = OFF_ENCS + ROWS*ZDIM*4;     // f32  16x64
constexpr unsigned OFF_PRIS = OFF_PRIM + ROWS*ZDIM*4;     // f32  16x64
constexpr unsigned OFF_DECM = OFF_PRIS + ROWS*ZDIM*4;     // f32  16x128
constexpr unsigned OFF_DECS = OFF_DECM + ROWS*XDIM*4;     // f32  16x128
constexpr unsigned OFF_GI   = OFF_DECS + ROWS*XDIM*4;     // f32  16x768
constexpr unsigned OFF_GH   = OFF_GI   + ROWS*3*HDIM*4;   // f32  16x768
constexpr unsigned OFF_RED  = OFF_GH   + ROWS*3*HDIM*4;   // f32  reduce slot
constexpr unsigned SMEM_BYTES = OFF_RED + 64;             // ~214 KB < 320 KB/WGP

// ---------------- WMMA fragment helpers (ISA 7.12.2 layouts) ----------------
union AB16 { v16bf v; v8bf h[2]; };

// A: 16x32 bf16, row-major in LDS, pitch elements. lane: row = lane&15,
// half = lane>>4. half0 -> K {0..7,16..23}, half1 -> K {8..15,24..31}.
__device__ __forceinline__ v16bf load_a_frag(const bf16* A, int pitch, int r,
                                             int half, int k0) {
  const bf16* p = A + r * pitch + k0 + half * 8;
  AB16 u;
  u.h[0] = *(const v8bf*)(p);
  u.h[1] = *(const v8bf*)(p + 16);
  return u.v;
}

// B: 32x16 (KxN). lane holds column n = lane&15, half0 -> K {0..7,16..23},
// half1 -> K {8..15,24..31}, from the row-major PyTorch weight W[n][k].
__device__ __forceinline__ v16bf load_b_frag(const bf16* wrow, int half, int k0) {
  const bf16* p = wrow + k0 + half * 8;
  AB16 u;
  u.h[0] = *(const v8bf*)(p);
  u.h[1] = *(const v8bf*)(p + 16);
  return u.v;
}

__device__ __forceinline__ float act_relu(float v)     { return v > 0.f ? v : 0.f; }
__device__ __forceinline__ float act_softplus(float v) { return v > 20.f ? v : log1pf(__expf(v)); }
__device__ __forceinline__ float act_sigmoid(float v)  { return 1.f / (1.f + __expf(-v)); }

// GEMM: [16 x (K0+K1)] @ W^T -> [16 x N]. A0/A1 are the (optionally
// concatenated) LDS activation operands; W is bf16 global [N][K0+K1].
// ACT: 0 none, 1 relu, 2 softplus, 3 sigmoid.
template <int ACT>
__device__ __forceinline__ void wg_gemm(const bf16* A0, int K0,
                                        const bf16* A1, int K1,
                                        const bf16* __restrict__ W,
                                        const float* __restrict__ bias, int N,
                                        bf16* outB, float* outF, int outPitch) {
  const int tid  = threadIdx.x;
  const int wave = tid >> 5;
  const int lane = tid & 31;
  const int half = lane >> 4;
  const int r    = lane & 15;
  const int Ktot = K0 + K1;
  const int nTiles = N >> 4;
  for (int nt = wave; nt < nTiles; nt += NWAVES) {
    const int col = (nt << 4) + r;
    const bf16* wrow = W + (size_t)col * Ktot;
    if (nt + NWAVES < nTiles)  // global_prefetch_b8 next weight row for this wave
      __builtin_prefetch(W + (size_t)(col + (NWAVES << 4)) * Ktot, 0, 1);
    v8f acc = {};
    for (int k = 0; k < K0; k += 32) {
      v16bf a = load_a_frag(A0, K0, r, half, k);
      v16bf b = load_b_frag(wrow, half, k);
      acc = __builtin_amdgcn_wmma_f32_16x16x32_bf16(false, a, false, b,
                                                    (short)0, acc, false, false);
    }
    for (int k = 0; k < K1; k += 32) {
      v16bf a = load_a_frag(A1, K1, r, half, k);
      v16bf b = load_b_frag(wrow, half, K0 + k);
      acc = __builtin_amdgcn_wmma_f32_16x16x32_bf16(false, a, false, b,
                                                    (short)0, acc, false, false);
    }
    const float bv = bias[col];
    const int rowBase = half << 3;  // lane<16 -> M 0..7, lane>=16 -> M 8..15
#pragma unroll
    for (int i = 0; i < 8; ++i) {
      float v = acc[i] + bv;
      if (ACT == 1)      v = act_relu(v);
      else if (ACT == 2) v = act_softplus(v);
      else if (ACT == 3) v = act_sigmoid(v);
      const int m = rowBase + i;
      if (outB) outB[m * outPitch + col] = (bf16)v;
      if (outF) outF[m * outPitch + col] = v;
    }
  }
}

// ---------------- parameters ----------------
struct VParams {
  const float* x;
  const float* eps;
  const bf16 *w_phi1, *w_phi2, *w_phiz, *w_enc1, *w_enc2, *w_encm, *w_encs,
             *w_pri,  *w_prim, *w_pris, *w_dec1, *w_dec2, *w_decm, *w_decs,
             *w_gih,  *w_ghh;
  const float *b_phi1, *b_phi2, *b_phiz, *b_enc1, *b_enc2, *b_encm, *b_encs,
              *b_pri,  *b_prim, *b_pris, *b_dec1, *b_dec2, *b_decm, *b_decs,
              *b_gih,  *b_ghh;
  float* nllT;     // [T_STEPS] per-step NLL partial sums (pre-sigmoid)
  float* kldPart;  // single global KLD accumulator
};

// ---------------- persistent scan kernel: 1 WG = 16 batch rows, all T ----------------
__global__ __launch_bounds__(NTHREADS) void vrnn_scan_kernel(VParams p) {
  extern __shared__ __align__(16) char smem[];
  float* xF   = (float*)(smem + OFF_XF);
  bf16*  xB   = (bf16*) (smem + OFF_XB);
  float* hF   = (float*)(smem + OFF_HF);
  bf16*  hB   = (bf16*) (smem + OFF_HB);
  bf16*  t0   = (bf16*) (smem + OFF_T0);
  bf16*  phix = (bf16*) (smem + OFF_PHIX);
  bf16*  ench = (bf16*) (smem + OFF_ENCH);
  bf16*  prih = (bf16*) (smem + OFF_PRIH);
  bf16*  dech = (bf16*) (smem + OFF_DECH);
  bf16*  phiz = (bf16*) (smem + OFF_PHIZ);
  bf16*  zB   = (bf16*) (smem + OFF_ZB);
  float* encm = (float*)(smem + OFF_ENCM);
  float* encs = (float*)(smem + OFF_ENCS);
  float* prim = (float*)(smem + OFF_PRIM);
  float* pris = (float*)(smem + OFF_PRIS);
  float* decm = (float*)(smem + OFF_DECM);
  float* decs = (float*)(smem + OFF_DECS);
  float* gi   = (float*)(smem + OFF_GI);
  float* gh   = (float*)(smem + OFF_GH);
  float* red  = (float*)(smem + OFF_RED);

  const int tid  = threadIdx.x;
  const int row0 = blockIdx.x * ROWS;

  for (int i = tid; i < ROWS * HDIM; i += NTHREADS) { hF[i] = 0.f; hB[i] = (bf16)0.f; }
  float kldAcc = 0.f;
  __syncthreads();

  for (int t = 0; t < T_STEPS; ++t) {
    // load x_t slice (rows contiguous in batch-major layout)
    const float* xt = p.x + ((size_t)t * BATCH + row0) * XDIM;
    for (int i = tid; i < ROWS * XDIM; i += NTHREADS) {
      float v = xt[i]; xF[i] = v; xB[i] = (bf16)v;
    }
    __syncthreads();

    wg_gemm<1>(xB,  XDIM, nullptr, 0, p.w_phi1, p.b_phi1, HDIM, t0,   nullptr, HDIM);
    __syncthreads();
    wg_gemm<1>(t0,  HDIM, nullptr, 0, p.w_phi2, p.b_phi2, HDIM, phix, nullptr, HDIM);
    __syncthreads();
    wg_gemm<1>(phix, HDIM, hB, HDIM, p.w_enc1, p.b_enc1, HDIM, t0,   nullptr, HDIM);
    __syncthreads();
    wg_gemm<1>(t0,  HDIM, nullptr, 0, p.w_enc2, p.b_enc2, HDIM, ench, nullptr, HDIM);
    __syncthreads();
    wg_gemm<0>(ench, HDIM, nullptr, 0, p.w_encm, p.b_encm, ZDIM, nullptr, encm, ZDIM);
    wg_gemm<2>(ench, HDIM, nullptr, 0, p.w_encs, p.b_encs, ZDIM, nullptr, encs, ZDIM);
    wg_gemm<1>(hB,  HDIM, nullptr, 0, p.w_pri,  p.b_pri,  HDIM, prih, nullptr, HDIM);
    __syncthreads();
    wg_gemm<0>(prih, HDIM, nullptr, 0, p.w_prim, p.b_prim, ZDIM, nullptr, prim, ZDIM);
    wg_gemm<2>(prih, HDIM, nullptr, 0, p.w_pris, p.b_pris, ZDIM, nullptr, pris, ZDIM);
    __syncthreads();

    // reparameterize z (enc_std treated as variance, faithful) + KLD accumulation
    const float* ep = p.eps + ((size_t)t * BATCH + row0) * ZDIM;
    for (int i = tid; i < ROWS * ZDIM; i += NTHREADS) {
      float em = encm[i], es = encs[i], pm = prim[i], ps = pris[i];
      float zv = ep[i] * sqrtf(es) + em;
      zB[i] = (bf16)zv;
      float dm = em - pm;
      kldAcc += 0.5f * (2.f * logf(ps + EPSF) - 2.f * logf(es + EPSF) +
                        (es * es + dm * dm) / (ps * ps) - 1.f);
    }
    __syncthreads();

    wg_gemm<1>(zB,  ZDIM, nullptr, 0, p.w_phiz, p.b_phiz, HDIM, phiz, nullptr, HDIM);
    __syncthreads();
    wg_gemm<1>(phiz, HDIM, hB, HDIM, p.w_dec1, p.b_dec1, HDIM, t0,   nullptr, HDIM);
    __syncthreads();
    wg_gemm<1>(t0,  HDIM, nullptr, 0, p.w_dec2, p.b_dec2, HDIM, dech, nullptr, HDIM);
    __syncthreads();
    wg_gemm<3>(dech, HDIM, nullptr, 0, p.w_decm, p.b_decm, XDIM,   nullptr, decm, XDIM);
    wg_gemm<2>(dech, HDIM, nullptr, 0, p.w_decs, p.b_decs, XDIM,   nullptr, decs, XDIM);
    wg_gemm<0>(phix, HDIM, phiz, HDIM, p.w_gih, p.b_gih, 3*HDIM,   nullptr, gi,   3*HDIM);
    wg_gemm<0>(hB,  HDIM, nullptr, 0, p.w_ghh, p.b_ghh, 3*HDIM,    nullptr, gh,   3*HDIM);
    __syncthreads();

    if (tid == 0) red[0] = 0.f;
    __syncthreads();

    // NLL partial (per-step; global sigmoid applied in finalize kernel)
    float np = 0.f;
    for (int i = tid; i < ROWS * XDIM; i += NTHREADS) {
      float dsd = decs[i], dmu = decm[i], xv = xF[i];
      float d = xv - dmu;
      np += logf(dsd + EPSF) + LOG_2PI_HALF + d * d / (2.f * dsd * dsd);
    }
    atomicAdd(&red[0], np);  // ds_add_f32 workgroup reduction

    // GRU cell (PyTorch gate order r, z, n)
    for (int i = tid; i < ROWS * HDIM; i += NTHREADS) {
      const int m = i >> 8, n = i & (HDIM - 1);
      const float* gim = gi + m * 3 * HDIM;
      const float* ghm = gh + m * 3 * HDIM;
      float rg = act_sigmoid(gim[n] + ghm[n]);
      float zg = act_sigmoid(gim[HDIM + n] + ghm[HDIM + n]);
      float ng = tanhf(gim[2 * HDIM + n] + rg * ghm[2 * HDIM + n]);
      float hn = (1.f - zg) * ng + zg * hF[i];
      hF[i] = hn; hB[i] = (bf16)hn;
    }
    __syncthreads();
    if (tid == 0) atomicAdd(&p.nllT[t], red[0]);
    __syncthreads();
  }

  // KLD: workgroup reduce then one global atomic
  if (tid == 0) red[0] = 0.f;
  __syncthreads();
  atomicAdd(&red[0], kldAcc);
  __syncthreads();
  if (tid == 0) atomicAdd(p.kldPart, red[0]);
}

// ---------------- utility kernels ----------------
__global__ void zero_f32_kernel(float* p, int n) {
  int i = blockIdx.x * blockDim.x + threadIdx.x;
  if (i < n) p[i] = 0.f;
}

__global__ void cvt_f32_to_bf16_kernel(const float* __restrict__ s,
                                       bf16* __restrict__ d, int n) {
  int i = blockIdx.x * blockDim.x + threadIdx.x;
  if (i < n) d[i] = (bf16)s[i];
}

__global__ void vrnn_finalize_kernel(const float* __restrict__ nllT,
                                     const float* __restrict__ kldPart,
                                     float* __restrict__ out) {
  __shared__ float red;
  if (threadIdx.x == 0) red = 0.f;
  __syncthreads();
  float s = 0.f;
  for (int t = threadIdx.x; t < T_STEPS; t += blockDim.x)
    s += 1.f / (1.f + __expf(-nllT[t]));
  atomicAdd(&red, s);
  __syncthreads();
  if (threadIdx.x == 0) { out[0] = kldPart[0]; out[1] = red; }
}

// ---------------- host launcher ----------------
extern "C" void kernel_launch(void* const* d_in, const int* in_sizes, int n_in,
                              void* d_out, int out_size, void* d_ws, size_t ws_size,
                              hipStream_t stream) {
  (void)in_sizes; (void)n_in; (void)out_size; (void)ws_size;

  // d_in order: 0 x, 1 eps, then params dict in insertion order (W,b pairs).
  static const int widx[16] = {2, 4, 6, 8, 10, 12, 14, 16, 18, 20, 22, 24, 26, 28, 30, 32};
  static const int bidx[16] = {3, 5, 7, 9, 11, 13, 15, 17, 19, 21, 23, 25, 27, 29, 31, 33};
  static const int wsz[16] = {
      HDIM * XDIM,      // phi_x_w1
      HDIM * HDIM,      // phi_x_w2
      HDIM * ZDIM,      // phi_z_w
      HDIM * 2 * HDIM,  // enc_w1
      HDIM * HDIM,      // enc_w2
      ZDIM * HDIM,      // enc_mean_w
      ZDIM * HDIM,      // enc_std_w
      HDIM * HDIM,      // prior_w
      ZDIM * HDIM,      // prior_mean_w
      ZDIM * HDIM,      // prior_std_w
      HDIM * 2 * HDIM,  // dec_w1
      HDIM * HDIM,      // dec_w2
      XDIM * HDIM,      // dec_mean_w
      XDIM * HDIM,      // dec_std_w
      3 * HDIM * 2 * HDIM,  // gru_w_ih
      3 * HDIM * HDIM       // gru_w_hh
  };

  char* ws = (char*)d_ws;
  size_t off = 0;
  auto carve = [&](size_t bytes) -> char* {
    char* p = ws + off;
    off = (off + bytes + 255) & ~(size_t)255;
    return p;
  };
  float* kldPart = (float*)carve(sizeof(float));
  float* nllT    = (float*)carve(T_STEPS * sizeof(float));
  const int zeroFloats = (int)(off / 4);  // accumulators region
  bf16* Wb[16];
  for (int i = 0; i < 16; ++i) Wb[i] = (bf16*)carve((size_t)wsz[i] * sizeof(bf16));

  // 1) zero the accumulators (harness poisons ws with 0xAA)
  zero_f32_kernel<<<(zeroFloats + 255) / 256, 256, 0, stream>>>((float*)d_ws, zeroFloats);

  // 2) convert fp32 weights -> bf16 in workspace (stay resident in L2)
  for (int i = 0; i < 16; ++i) {
    int n = wsz[i];
    cvt_f32_to_bf16_kernel<<<(n + 255) / 256, 256, 0, stream>>>(
        (const float*)d_in[widx[i]], Wb[i], n);
  }

  // 3) persistent scan: 16 workgroups x 256 threads, ~214 KB dynamic LDS each
  VParams p;
  p.x   = (const float*)d_in[0];
  p.eps = (const float*)d_in[1];
  p.w_phi1 = Wb[0];  p.w_phi2 = Wb[1];  p.w_phiz = Wb[2];  p.w_enc1 = Wb[3];
  p.w_enc2 = Wb[4];  p.w_encm = Wb[5];  p.w_encs = Wb[6];  p.w_pri  = Wb[7];
  p.w_prim = Wb[8];  p.w_pris = Wb[9];  p.w_dec1 = Wb[10]; p.w_dec2 = Wb[11];
  p.w_decm = Wb[12]; p.w_decs = Wb[13]; p.w_gih  = Wb[14]; p.w_ghh  = Wb[15];
  p.b_phi1 = (const float*)d_in[bidx[0]];  p.b_phi2 = (const float*)d_in[bidx[1]];
  p.b_phiz = (const float*)d_in[bidx[2]];  p.b_enc1 = (const float*)d_in[bidx[3]];
  p.b_enc2 = (const float*)d_in[bidx[4]];  p.b_encm = (const float*)d_in[bidx[5]];
  p.b_encs = (const float*)d_in[bidx[6]];  p.b_pri  = (const float*)d_in[bidx[7]];
  p.b_prim = (const float*)d_in[bidx[8]];  p.b_pris = (const float*)d_in[bidx[9]];
  p.b_dec1 = (const float*)d_in[bidx[10]]; p.b_dec2 = (const float*)d_in[bidx[11]];
  p.b_decm = (const float*)d_in[bidx[12]]; p.b_decs = (const float*)d_in[bidx[13]];
  p.b_gih  = (const float*)d_in[bidx[14]]; p.b_ghh  = (const float*)d_in[bidx[15]];
  p.nllT = nllT;
  p.kldPart = kldPart;

  vrnn_scan_kernel<<<NWG, NTHREADS, SMEM_BYTES, stream>>>(p);

  // 4) finalize: kld = sum, nll = sum_t sigmoid(nll_t)
  vrnn_finalize_kernel<<<1, 256, 0, stream>>>(nllT, kldPart, (float*)d_out);
}
